// AdjacencyPolicyModule_5257039970311
// MI455X (gfx1250) — matrix operand order repaired and checked by reading, not compiled, not run on hardware
//
#include <hip/hip_runtime.h>
#include <hip/hip_bf16.h>
#include <stdint.h>

#define B_   4096
#define N_   128
#define D_   256
#define EOUT 16256
#define OUTW 16257   // EOUT + 1 (exit action)

typedef __attribute__((ext_vector_type(16))) __bf16 v16bf;
typedef __attribute__((ext_vector_type(8)))  float  v8f;

union Frag16 { v16bf v; uint4 q[2]; };

__device__ __forceinline__ unsigned short f2bf(float f) {
    unsigned int u = __float_as_uint(f);
    u = u + 0x7FFFu + ((u >> 16) & 1u);      // round-to-nearest-even
    return (unsigned short)(u >> 16);
}
__device__ __forceinline__ float bf2f(unsigned short h) {
    return __uint_as_float(((unsigned int)h) << 16);
}

// A fragment: 16x32 bf16, matrix row-major [rows][ldk], per ISA 7.12.2:
// lane<16: row=lane, K = kb+{0..7} (VGPR0-3) and kb+{16..23} (VGPR4-7)
// lane>=16: row=lane-16, K = kb+{8..15} and kb+{24..31}
__device__ __forceinline__ v16bf load_fragA(const unsigned short* base, int row,
                                            int ldk, int kb, int lane) {
    int r = row + (lane & 15);
    int k = kb + ((lane >> 4) << 3);
    Frag16 f;
    f.q[0] = *(const uint4*)(base + (size_t)r * ldk + k);
    f.q[1] = *(const uint4*)(base + (size_t)r * ldk + k + 16);
    return f.v;
}

// B fragment: 32x16 bf16 (KxN), weights pre-transposed to [N][K] row-major.
// lane<16: col=lane, K = kb+{0..7} and kb+{16..23}; lane>=16: col=lane-16, K = kb+{8..15} and kb+{24..31}
__device__ __forceinline__ v16bf load_fragB(const unsigned short* base, int col,
                                            int ldk, int kb, int lane) {
    int c = col + (lane & 15);
    int k = kb + ((lane >> 4) << 3);
    Frag16 f;
    f.q[0] = *(const uint4*)(base + (size_t)c * ldk + k);
    f.q[1] = *(const uint4*)(base + (size_t)c * ldk + k + 16);
    return f.v;
}

__device__ __forceinline__ v8f wmma_bf16(v16bf a, v16bf b, v8f c) {
    return __builtin_amdgcn_wmma_f32_16x16x32_bf16(false, a, false, b,
                                                   (short)0, c, false, false);
}

__device__ __forceinline__ int lbound(const int* a, int n, int key) {
    int lo = 0, hi = n;
    while (lo < hi) { int mid = (lo + hi) >> 1; if (a[mid] < key) lo = mid + 1; else hi = mid; }
    return lo;
}

// ---------------- K1: fp32 [K][Nn] -> bf16 transposed [Nn][K] ----------------
__global__ __launch_bounds__(256) void convT(const float* __restrict__ in,
                                             unsigned short* __restrict__ outp,
                                             int K, int Nn) {
    int idx = blockIdx.x * 256 + threadIdx.x;
    if (idx >= K * Nn) return;
    int n = idx / K, k = idx - n * K;
    outp[idx] = f2bf(in[(size_t)k * Nn + n]);
}

// ---------------- K2: layer0 = relu(LN(x@W0+b0)) via edge gather -------------
// One block per graph; LDS bitmap dedupes edges (reference .set(1.0) semantics).
__global__ __launch_bounds__(256) void layer0_gather(
    const int* __restrict__ esrc, const int* __restrict__ edst,
    const int* __restrict__ ebat, const float* __restrict__ W0,
    const float* __restrict__ b0, const float* __restrict__ g0,
    const float* __restrict__ be0, unsigned short* __restrict__ h0, int E) {
    __shared__ unsigned int bitmap[512];   // N*N = 16384 bits
    __shared__ unsigned int list[1024];
    __shared__ unsigned int cnt;
    __shared__ float red[16];

    int b = blockIdx.x, tid = threadIdx.x;
    int lo = lbound(ebat, E, b);
    int hi = lbound(ebat, E, b + 1);

    bitmap[tid] = 0; bitmap[tid + 256] = 0;
    if (tid == 0) cnt = 0;
    __syncthreads();

    for (int e = lo + tid; e < hi; e += 256) {
        int idx = esrc[e] * N_ + edst[e];
        unsigned int w = (unsigned)idx >> 5, m = 1u << (idx & 31);
        unsigned int old = atomicOr(&bitmap[w], m);
        if (!(old & m)) {
            unsigned int p = atomicAdd(&cnt, 1u);
            if (p < 1024u) list[p] = (unsigned)idx;
        }
    }
    __syncthreads();
    int nE = (int)min(cnt, 1024u);

    float acc = b0[tid];
    for (int i = 0; i < nE; ++i)
        acc += W0[(size_t)list[i] * D_ + tid];   // LDS broadcast of list[i]

    // LayerNorm across 256 channels
    float s = acc, s2 = acc * acc;
    for (int off = 16; off; off >>= 1) {
        s  += __shfl_xor(s,  off, 32);
        s2 += __shfl_xor(s2, off, 32);
    }
    int wave = tid >> 5, lane = tid & 31;
    if (lane == 0) { red[wave] = s; red[8 + wave] = s2; }
    __syncthreads();
    float S = 0.f, S2 = 0.f;
    for (int i = 0; i < 8; ++i) { S += red[i]; S2 += red[8 + i]; }
    float mean = S * (1.f / 256.f);
    float var  = S2 * (1.f / 256.f) - mean * mean;
    float y = (acc - mean) * rsqrtf(var + 1e-5f) * g0[tid] + be0[tid];
    h0[(size_t)b * D_ + tid] = f2bf(fmaxf(y, 0.f));
}

// ------- K3: Y[B,256] = epilogue(X[B,256] @ W[256,256] + bias) via WMMA ------
// 512 threads = 16 waves; wave w -> 16x16 tile at cols [16w,16w+16); block = 16 rows.
// doLN=1: relu(LayerNorm(.)); doLN=0: bias only.
__global__ __launch_bounds__(512) void gemm256_fused(
    const unsigned short* __restrict__ X, const unsigned short* __restrict__ Wt,
    const float* __restrict__ bias, const float* __restrict__ gamma,
    const float* __restrict__ beta, unsigned short* __restrict__ Y, int doLN) {
    __shared__ float tile[16][D_ + 8];

    int wave = threadIdx.x >> 5, lane = threadIdx.x & 31;
    int rowBase = blockIdx.x * 16;
    int colBase = wave * 16;

    v8f acc = {};
    #pragma unroll
    for (int kb = 0; kb < D_; kb += 32) {
        v16bf a = load_fragA(X,  rowBase, D_, kb, lane);
        v16bf bb = load_fragB(Wt, colBase, D_, kb, lane);
        acc = wmma_bf16(a, bb, acc);
    }
    int n  = colBase + (lane & 15);
    int mb = (lane >> 4) << 3;
    #pragma unroll
    for (int r = 0; r < 8; ++r) tile[mb + r][n] = acc[r];
    __syncthreads();

    int row = wave;                 // 16 waves handle 16 rows
    float vals[8], s = 0.f, s2 = 0.f;
    #pragma unroll
    for (int j = 0; j < 8; ++j) {
        int c = lane * 8 + j;
        float v = tile[row][c] + bias[c];
        vals[j] = v; s += v; s2 += v * v;
    }
    for (int off = 16; off; off >>= 1) {
        s  += __shfl_xor(s,  off, 32);
        s2 += __shfl_xor(s2, off, 32);
    }
    size_t gRow = (size_t)(rowBase + row) * D_;
    if (doLN) {
        float mean = s * (1.f / 256.f);
        float rinv = rsqrtf(s2 * (1.f / 256.f) - mean * mean + 1e-5f);
        #pragma unroll
        for (int j = 0; j < 8; ++j) {
            int c = lane * 8 + j;
            float y = (vals[j] - mean) * rinv * gamma[c] + beta[c];
            Y[gRow + c] = f2bf(fmaxf(y, 0.f));
        }
    } else {
        #pragma unroll
        for (int j = 0; j < 8; ++j) Y[gRow + lane * 8 + j] = f2bf(vals[j]);
    }
}

// ------- K4: out[B,16256] = he @ We1 + beb1 (writes into stride-16257 rows) --
// 256 threads = 8 waves as 4x2; block tile 64 rows x 64 cols; K=256.
__global__ __launch_bounds__(256) void gemm_edge(
    const unsigned short* __restrict__ X, const unsigned short* __restrict__ Wt,
    const float* __restrict__ bias, float* __restrict__ out) {
    int wave = threadIdx.x >> 5, lane = threadIdx.x & 31;
    int rowBase = blockIdx.y * 64 + (wave >> 1) * 16;
    int colBase = blockIdx.x * 64 + (wave & 1) * 32;

    v8f acc0 = {}, acc1 = {};
    #pragma unroll
    for (int kb = 0; kb < D_; kb += 32) {
        v16bf a  = load_fragA(X,  rowBase,      D_, kb, lane);
        v16bf b0 = load_fragB(Wt, colBase,      D_, kb, lane);
        v16bf b1 = load_fragB(Wt, colBase + 16, D_, kb, lane);
        acc0 = wmma_bf16(a, b0, acc0);
        acc1 = wmma_bf16(a, b1, acc1);
    }
    int n0 = colBase + (lane & 15);
    int m0 = rowBase + ((lane >> 4) << 3);
    float bi0 = bias[n0], bi1 = bias[n0 + 16];
    #pragma unroll
    for (int r = 0; r < 8; ++r) {
        size_t o = (size_t)(m0 + r) * OUTW + n0;
        out[o]      = acc0[r] + bi0;
        out[o + 16] = acc1[r] + bi1;
    }
}

// ---------------- K5: exit head: out[b,16256] = hx[b,:]·Wx1 + bx1 ------------
__global__ __launch_bounds__(256) void exit_head(
    const unsigned short* __restrict__ hx, const float* __restrict__ Wx1,
    const float* __restrict__ bx1, float* __restrict__ out) {
    __shared__ float red[8];
    int b = blockIdx.x, t = threadIdx.x;
    float v = bf2f(hx[(size_t)b * D_ + t]) * Wx1[t];
    for (int off = 16; off; off >>= 1) v += __shfl_xor(v, off, 32);
    if ((t & 31) == 0) red[t >> 5] = v;
    __syncthreads();
    if (t == 0) {
        float s = 0.f;
        for (int i = 0; i < 8; ++i) s += red[i];
        out[(size_t)b * OUTW + EOUT] = s + bx1[0];
    }
}

extern "C" void kernel_launch(void* const* d_in, const int* in_sizes, int n_in,
                              void* d_out, int out_size, void* d_ws, size_t ws_size,
                              hipStream_t stream) {
    const int*   esrc = (const int*)d_in[0];
    const int*   edst = (const int*)d_in[1];
    const int*   ebat = (const int*)d_in[2];
    const float* W0   = (const float*)d_in[3];
    const float* b0   = (const float*)d_in[4];
    const float* g0   = (const float*)d_in[5];
    const float* be0  = (const float*)d_in[6];
    const float* W1   = (const float*)d_in[7];
    const float* b1   = (const float*)d_in[8];
    const float* g1   = (const float*)d_in[9];
    const float* be1  = (const float*)d_in[10];
    const float* Wm   = (const float*)d_in[11];
    const float* bm   = (const float*)d_in[12];
    const float* We0  = (const float*)d_in[13];
    const float* beb0 = (const float*)d_in[14];
    const float* ge0  = (const float*)d_in[15];
    const float* bee0 = (const float*)d_in[16];
    const float* We1  = (const float*)d_in[17];
    const float* beb1 = (const float*)d_in[18];
    const float* Wx0  = (const float*)d_in[19];
    const float* bx0  = (const float*)d_in[20];
    const float* gx0  = (const float*)d_in[21];
    const float* bxe0 = (const float*)d_in[22];
    const float* Wx1  = (const float*)d_in[23];
    const float* bx1  = (const float*)d_in[24];
    float* out = (float*)d_out;
    const int E = in_sizes[0];

    // workspace carve (all 256B-aligned chunk sizes)
    char* ws = (char*)d_ws;
    unsigned short* h0   = (unsigned short*)ws; ws += (size_t)B_ * D_ * 2;
    unsigned short* h1   = (unsigned short*)ws; ws += (size_t)B_ * D_ * 2;
    unsigned short* emb  = (unsigned short*)ws; ws += (size_t)B_ * D_ * 2;
    unsigned short* he   = (unsigned short*)ws; ws += (size_t)B_ * D_ * 2;
    unsigned short* hx   = (unsigned short*)ws; ws += (size_t)B_ * D_ * 2;
    unsigned short* W1t  = (unsigned short*)ws; ws += (size_t)D_ * D_ * 2;
    unsigned short* Wmt  = (unsigned short*)ws; ws += (size_t)D_ * D_ * 2;
    unsigned short* We0t = (unsigned short*)ws; ws += (size_t)D_ * D_ * 2;
    unsigned short* Wx0t = (unsigned short*)ws; ws += (size_t)D_ * D_ * 2;
    unsigned short* We1t = (unsigned short*)ws; ws += (size_t)EOUT * D_ * 2;

    // K1: weight transpose+bf16
    int nSq = D_ * D_;
    convT<<<(nSq + 255) / 256, 256, 0, stream>>>(W1,  W1t,  D_, D_);
    convT<<<(nSq + 255) / 256, 256, 0, stream>>>(Wm,  Wmt,  D_, D_);
    convT<<<(nSq + 255) / 256, 256, 0, stream>>>(We0, We0t, D_, D_);
    convT<<<(nSq + 255) / 256, 256, 0, stream>>>(Wx0, Wx0t, D_, D_);
    int nE1 = D_ * EOUT;
    convT<<<(nE1 + 255) / 256, 256, 0, stream>>>(We1, We1t, D_, EOUT);

    // K2: sparse adjacency -> layer0
    layer0_gather<<<B_, 256, 0, stream>>>(esrc, edst, ebat, W0, b0, g0, be0, h0, E);

    // K3: fused small GEMMs
    gemm256_fused<<<B_ / 16, 512, 0, stream>>>(h0,  W1t,  b1,   g1,  be1,  h1,  1);
    gemm256_fused<<<B_ / 16, 512, 0, stream>>>(h1,  Wmt,  bm,   bm,  bm,   emb, 0);
    gemm256_fused<<<B_ / 16, 512, 0, stream>>>(emb, We0t, beb0, ge0, bee0, he,  1);
    gemm256_fused<<<B_ / 16, 512, 0, stream>>>(emb, Wx0t, bx0,  gx0, bxe0, hx,  1);

    // K4: dominant GEMM -> edge_actions
    dim3 gE(EOUT / 64, B_ / 64);
    gemm_edge<<<gE, 256, 0, stream>>>(he, We1t, beb1, out);

    // K5: exit action
    exit_head<<<B_, 256, 0, stream>>>(hx, Wx1, bx1, out);
}
